// HierarchicalPathScorer_69741678952869
// MI455X (gfx1250) — matrix operand order repaired and checked by reading, not compiled, not run on hardware
//
#include <hip/hip_runtime.h>
#include <math.h>

typedef __attribute__((ext_vector_type(16))) _Float16 v16h;
typedef __attribute__((ext_vector_type(8)))  float    v8f;

#define BASE_F 7.0f

// ---------------- helpers ----------------

__device__ __forceinline__ float pnormf(float x, float g, float b) {
  // jnp.mod(x, 7): result has sign of divisor
  float m = x - floorf(x * (1.0f / BASE_F)) * BASE_F;
  return g * ((m - 0.5f * BASE_F) * (2.0f / BASE_F)) + b;
}

// Build a 16x32 f16 A-fragment from an f32 matrix in LDS.
// Element (M, K) lives at src[K * stride + M]; rows M >= nvalid are zero.
// This code's contiguous K order (B packed to match): lanes 0-15 hold M=lane; halves 0..7 are
// K = kb+0..7 (kb = 16 for lanes 16-31), halves 8..15 are K = kb+8..15.
__device__ __forceinline__ v16h hps_build_a(const float* __restrict__ src,
                                            int stride, int nvalid,
                                            int k0, int lane) {
  v16h a;
  int m  = lane & 15;
  int kb = (lane >> 4) * 16;
  if (m < nvalid) {
#pragma unroll
    for (int i = 0; i < 8; ++i)
      a[i] = (_Float16)src[(k0 + kb + i) * stride + m];
#pragma unroll
    for (int i = 0; i < 8; ++i)
      a[8 + i] = (_Float16)src[(k0 + kb + 8 + i) * stride + m];
  } else {
#pragma unroll
    for (int i = 0; i < 16; ++i) a[i] = (_Float16)0.0f;
  }
  return a;
}

// ---------------- prep kernel: fold rotations, pack f16 B-fragments ----------------
// W1frag layout: idx = (((t*2+nt)*16+ks)*32 + lane)*16 + i
//   -> lane's 16 contiguous halves for WMMA B tile (n-tile nt, k-step ks) of team t
//   o = nt*16 + (lane&15), f = ks*32 + (lane>>4)*16 + i  (B 32x16: lanes=N col, K packed)
// mW1frag layout: idx = ((w*16+ks)*32 + lane)*16 + i, w = o-tile 0..3

__global__ void hps_prep(const float* __restrict__ W1,
                         const float* __restrict__ ang1,
                         const float* __restrict__ W2,
                         const float* __restrict__ ang2,
                         const float* __restrict__ mW1,
                         const float* __restrict__ mang1,
                         const float* __restrict__ mW2,
                         const float* __restrict__ mang2,
                         _Float16* __restrict__ W1frag,
                         _Float16* __restrict__ mW1frag,
                         float* __restrict__ W2r,
                         float* __restrict__ mW2r) {
  int idx = blockIdx.x * blockDim.x + threadIdx.x;

  if (idx < 9 * 2 * 16 * 512) {
    int i    = idx & 15;
    int lane = (idx >> 4) & 31;
    int ks   = (idx >> 9) & 15;
    int nt   = (idx >> 13) & 1;
    int t    = idx >> 14;
    int o = nt * 16 + (lane & 15);
    int f = ks * 32 + (lane >> 4) * 16 + i;
    const float* w = W1 + (t * 32 + o) * 512;
    float v;
    if (f >= 4) {
      v = w[f];
    } else {
      int p = f >> 1;
      float a = ang1[t * 2 + p];
      float c = cosf(a), s = sinf(a);
      float w0 = w[2 * p], w1 = w[2 * p + 1];
      v = (f & 1) ? (-s * w0 + c * w1) : (c * w0 + s * w1);
    }
    W1frag[idx] = (_Float16)v;
  }

  if (idx < 4 * 16 * 512) {
    int i    = idx & 15;
    int lane = (idx >> 4) & 31;
    int ks   = (idx >> 9) & 15;
    int wt   = idx >> 13;  // o-tile 0..3
    int o = wt * 16 + (lane & 15);
    int f = ks * 32 + (lane >> 4) * 16 + i;
    const float* w = mW1 + o * 512;
    float v;
    if (f >= 8) {
      v = w[f];
    } else {
      int p = f >> 1;
      float a = mang1[p];
      float c = cosf(a), s = sinf(a);
      float w0 = w[2 * p], w1 = w[2 * p + 1];
      v = (f & 1) ? (-s * w0 + c * w1) : (c * w0 + s * w1);
    }
    mW1frag[idx] = (_Float16)v;
  }

  if (idx < 288) {  // W2 with ang2 folded: scores = pnorm(h) . W2r
    int t = idx / 32, o = idx % 32;
    const float* w = W2 + t * 32;
    float v;
    if (o >= 4) {
      v = w[o];
    } else {
      int p = o >> 1;
      float a = ang2[t * 2 + p];
      float c = cosf(a), s = sinf(a);
      float w0 = w[2 * p], w1 = w[2 * p + 1];
      v = (o & 1) ? (-s * w0 + c * w1) : (c * w0 + s * w1);
    }
    W2r[idx] = v;
  }

  if (idx < 64) {  // mW2 with mang2 folded
    int o = idx;
    float v;
    if (o >= 4) {
      v = mW2[o];
    } else {
      int p = o >> 1;
      float a = mang2[p];
      float c = cosf(a), s = sinf(a);
      float w0 = mW2[2 * p], w1 = mW2[2 * p + 1];
      v = (o & 1) ? (-s * w0 + c * w1) : (c * w0 + s * w1);
    }
    mW2r[idx] = v;
  }
}

// ---------------- main kernel: one workgroup (9 waves) per batch ----------------

__global__ __launch_bounds__(288)
void hps_main(const float* __restrict__ path,
              const _Float16* __restrict__ W1frag,
              const float* __restrict__ b1v,
              const float* __restrict__ gam1,
              const float* __restrict__ bet1,
              const float* __restrict__ W2r,
              const float* __restrict__ b2v,
              const _Float16* __restrict__ mW1frag,
              const float* __restrict__ mb1v,
              const float* __restrict__ mgam,
              const float* __restrict__ mbet,
              const float* __restrict__ mW2r,
              const float* __restrict__ mb2v,
              const float* __restrict__ comm_c,
              const float* __restrict__ res_c,
              const float* __restrict__ int_c,
              float* __restrict__ out) {
  extern __shared__ float smem[];
  float* xf      = smem;           // 512*81 = 41472 floats: x[f][p], p = t*9+s
  float* team    = xf + 41472;     // 512*9  = 4608:  team_out[f][t]
  float* scoresL = team + 4608;    // 81: scores[t][s]
  float* qL      = scoresL + 81;   // 81: q[t][s]
  float* tqL     = qL + 81;        // 9
  float* msPartL = tqL + 9;        // 36: [wave][t]
  float* mwL     = msPartL + 36;   // 9
  float* scaleL  = mwL + 9;        // 1

  const int tid  = threadIdx.x;
  const int lane = tid & 31;
  const int wv   = tid >> 5;   // 0..8
  const int b    = blockIdx.x;

  // ---- stage 1: stream this batch's x tile into LDS (read HBM exactly once) ----
  {
    const float4* src4 = (const float4*)(path + (size_t)b * (512 * 81));
    float4* dst4 = (float4*)xf;
    for (int i = tid; i < (512 * 81) / 4; i += 288) dst4[i] = src4[i];
  }
  __syncthreads();

  // ---- stage 2: per-team scorer GEMM (M=9pad16, K=512, N=32) via WMMA ----
  {
    const int t = wv;                 // each of the 9 waves owns one team
    v8f c0 = {};                      // N = 0..15
    v8f c1 = {};                      // N = 16..31
    const float* xs = xf + t * 9;     // element (s, f) at xs[f*81 + s]
    for (int ks = 0; ks < 16; ++ks) {
      v16h a = hps_build_a(xs, 81, 9, ks * 32, lane);
      const v16h* bp0 = (const v16h*)(W1frag + ((size_t)((t * 2 + 0) * 16 + ks)) * 512) + lane;
      const v16h* bp1 = (const v16h*)(W1frag + ((size_t)((t * 2 + 1) * 16 + ks)) * 512) + lane;
      v16h bm0 = *bp0;
      v16h bm1 = *bp1;
      c0 = __builtin_amdgcn_wmma_f32_16x16x32_f16(false, a, false, bm0, (short)0, c0, false, false);
      c1 = __builtin_amdgcn_wmma_f32_16x16x32_f16(false, a, false, bm1, (short)0, c1, false, false);
    }
    // C/D layout: lane holds (M = r + 8*(lane>>4), N = lane&15) in vgpr r
    int o0 = lane & 15, o1 = 16 + (lane & 15);
    float bi0 = b1v[t * 32 + o0],  bi1 = b1v[t * 32 + o1];
    float g0  = gam1[t * 32 + o0], g1  = gam1[t * 32 + o1];
    float e0  = bet1[t * 32 + o0], e1  = bet1[t * 32 + o1];
    float w20 = W2r[t * 32 + o0],  w21 = W2r[t * 32 + o1];
    float vr[8];
#pragma unroll
    for (int r = 0; r < 8; ++r) {
      vr[r] = pnormf(c0[r] + bi0, g0, e0) * w20 +
              pnormf(c1[r] + bi1, g1, e1) * w21;
    }
    // reduce across the 16 lanes of each half (covers all 32 o-channels)
#pragma unroll
    for (int off = 1; off <= 8; off <<= 1) {
#pragma unroll
      for (int r = 0; r < 8; ++r) vr[r] += __shfl_xor(vr[r], off, 32);
    }
    if (lane == 0) {
#pragma unroll
      for (int r = 0; r < 8; ++r) scoresL[t * 9 + r] = vr[r];  // s = 0..7
    }
    if (lane == 16) scoresL[t * 9 + 8] = vr[0];                // s = 8
  }
  __syncthreads();

  // ---- stage 3: softmax over s per team, plus tq ----
  if (tid < 9) {
    int t = tid;
    float b2s = b2v[t];
    float sc[9];
    float mx = -3.4e38f;
#pragma unroll
    for (int s = 0; s < 9; ++s) {
      sc[s] = scoresL[t * 9 + s] + b2s;
      mx = fmaxf(mx, sc[s]);
    }
    float sum = 0.f;
#pragma unroll
    for (int s = 0; s < 9; ++s) { sc[s] = expf(sc[s] - mx); sum += sc[s]; }
    float inv = 1.f / sum;
    float ts = 0.f;
#pragma unroll
    for (int s = 0; s < 9; ++s) {
      float qv = sc[s] * inv;
      qL[t * 9 + s] = qv;
      ts += qv;
    }
    tqL[t] = ts * (1.0f / 9.0f);
  }
  __syncthreads();

  // ---- stage 4: team_out[f][t] = sum_s x[t,s,f] * q[t,s] ----
  for (int idx = tid; idx < 9 * 512; idx += 288) {
    int t = idx % 9;
    int f = idx / 9;
    const float* xc = xf + f * 81 + t * 9;
    const float* qc = qL + t * 9;
    float acc = 0.f;
#pragma unroll
    for (int s = 0; s < 9; ++s) acc += xc[s] * qc[s];
    team[idx] = acc;
  }
  __syncthreads();

  // ---- stage 5: meta GEMM (M=9pad16, K=512, N=64) on waves 0-3; penalty on tid 287 ----
  if (wv < 4) {
    v8f cm = {};
    for (int ks = 0; ks < 16; ++ks) {
      v16h a = hps_build_a(team, 9, 9, ks * 32, lane);
      const v16h* bp = (const v16h*)(mW1frag + ((size_t)(wv * 16 + ks)) * 512) + lane;
      v16h bm = *bp;
      cm = __builtin_amdgcn_wmma_f32_16x16x32_f16(false, a, false, bm, (short)0, cm, false, false);
    }
    int o = wv * 16 + (lane & 15);
    float bi = mb1v[o], g = mgam[o], e = mbet[o], w2 = mW2r[o];
    float vr[8];
#pragma unroll
    for (int r = 0; r < 8; ++r) vr[r] = pnormf(cm[r] + bi, g, e) * w2;
#pragma unroll
    for (int off = 1; off <= 8; off <<= 1) {
#pragma unroll
      for (int r = 0; r < 8; ++r) vr[r] += __shfl_xor(vr[r], off, 32);
    }
    if (lane == 0) {
#pragma unroll
      for (int r = 0; r < 8; ++r) msPartL[wv * 9 + r] = vr[r];  // t = 0..7
    }
    if (lane == 16) msPartL[wv * 9 + 8] = vr[0];                // t = 8
  }
  if (tid == 287) {
    // tq == 1/9 mathematically (softmax rows sum to 1) -> computed faithfully
    float mean = 0.f;
#pragma unroll
    for (int t = 0; t < 9; ++t) mean += tqL[t];
    mean *= (1.0f / 9.0f);
    float var = 0.f;
#pragma unroll
    for (int t = 0; t < 9; ++t) { float d = tqL[t] - mean; var += d * d; }
    var *= (1.0f / 8.0f);  // ddof=1
    float nv = var / (mean * mean + 1e-8f);
    float total = comm_c[0] * 36.0f + res_c[0] * 9.0f + int_c[0] * 9.0f * (1.0f + nv);
    scaleL[0] = 1.0f - fminf(total, 0.5f);
  }
  __syncthreads();

  // ---- stage 6: meta softmax over t ----
  if (tid == 0) {
    float mb2s = mb2v[0];
    float ms[9];
    float mx = -3.4e38f;
#pragma unroll
    for (int t = 0; t < 9; ++t) {
      ms[t] = msPartL[t] + msPartL[9 + t] + msPartL[18 + t] + msPartL[27 + t] + mb2s;
      mx = fmaxf(mx, ms[t]);
    }
    float sum = 0.f;
#pragma unroll
    for (int t = 0; t < 9; ++t) { ms[t] = expf(ms[t] - mx); sum += ms[t]; }
    float inv = 1.f / sum;
#pragma unroll
    for (int t = 0; t < 9; ++t) mwL[t] = ms[t] * inv;
  }
  __syncthreads();

  // ---- stage 7: final[f] = (sum_t team_out[t,f] * mw[t]) * (1 - pen) ----
  float scl = scaleL[0];
  for (int f = tid; f < 512; f += 288) {
    const float* tc = team + f * 9;
    float acc = 0.f;
#pragma unroll
    for (int t = 0; t < 9; ++t) acc += tc[t] * mwL[t];
    out[(size_t)b * 512 + f] = acc * scl;
  }
}

// ---------------- launch ----------------

extern "C" void kernel_launch(void* const* d_in, const int* in_sizes, int n_in,
                              void* d_out, int out_size, void* d_ws, size_t ws_size,
                              hipStream_t stream) {
  (void)in_sizes; (void)n_in; (void)out_size; (void)ws_size;

  const float* path  = (const float*)d_in[0];
  const float* W1    = (const float*)d_in[1];
  const float* b1    = (const float*)d_in[2];
  const float* ang1  = (const float*)d_in[3];
  const float* gam1  = (const float*)d_in[4];
  const float* bet1  = (const float*)d_in[5];
  const float* W2    = (const float*)d_in[6];
  const float* b2    = (const float*)d_in[7];
  const float* ang2  = (const float*)d_in[8];
  const float* mW1   = (const float*)d_in[9];
  const float* mb1   = (const float*)d_in[10];
  const float* mang1 = (const float*)d_in[11];
  const float* mgam  = (const float*)d_in[12];
  const float* mbet  = (const float*)d_in[13];
  const float* mW2   = (const float*)d_in[14];
  const float* mb2   = (const float*)d_in[15];
  const float* mang2 = (const float*)d_in[16];
  const float* ccomm = (const float*)d_in[17];
  const float* cres  = (const float*)d_in[18];
  const float* cint  = (const float*)d_in[19];

  // workspace layout (bytes)
  char* ws = (char*)d_ws;
  _Float16* W1frag  = (_Float16*)(ws + 0);                 // 9*2*16*512 f16 = 294912 B
  _Float16* mW1frag = (_Float16*)(ws + 294912);            // 4*16*512  f16 =  65536 B
  float*    W2r     = (float*)(ws + 294912 + 65536);       // 288 f32
  float*    mW2r    = (float*)(ws + 294912 + 65536 + 288 * 4);  // 64 f32

  hps_prep<<<576, 256, 0, stream>>>(W1, ang1, W2, ang2, mW1, mang1, mW2, mang2,
                                    W1frag, mW1frag, W2r, mW2r);

  const size_t lds_bytes = (41472 + 4608 + 81 + 81 + 9 + 36 + 9 + 1) * sizeof(float);
  hps_main<<<4096, 288, lds_bytes, stream>>>(path, W1frag, b1, gam1, bet1, W2r, b2,
                                             mW1frag, mb1, mgam, mbet, mW2r, mb2,
                                             ccomm, cres, cint, (float*)d_out);
}
